// Approx_OTPE_67181878444197
// MI455X (gfx1250) — matrix operand order repaired and checked by reading, not compiled, not run on hardware
//
#include <hip/hip_runtime.h>
#include <hip/hip_bf16.h>

// ---------------------------------------------------------------------------
// SNN forward:  u' = sigmoid(2)*u + x_t @ W ; s = (u' >= 1) ; u = u' - s
//   (1) f32->f16 convert of X
//   (2) fused f32->f16 convert + transpose of W  ->  WT[N][K]
//   (3) one 4096x2048x2048 WMMA GEMM, double-buffered LDS, 8 wmma/K-chunk/wave
//   (4) per-(b,o) scan over T=32 (elementwise, state in regs)
// ---------------------------------------------------------------------------

typedef __attribute__((ext_vector_type(16))) _Float16 v16h;
typedef __attribute__((ext_vector_type(8)))  _Float16 v8h;
typedef __attribute__((ext_vector_type(4)))  _Float16 v4h;
typedef __attribute__((ext_vector_type(8)))  float    v8f;

#define T_STEPS 32
#define BATCH   128
#define IN_SZ   2048
#define OUT_SZ  2048
#define M_TOT   (T_STEPS * BATCH)      // 4096
#define SIG_TAU 0.8807970779778823f    // sigmoid(2.0)

// ---------------------------------------------------------------------------
// Kernel 1: f32 -> f16 conversion (vectorized x4)  -- for X
// ---------------------------------------------------------------------------
__global__ __launch_bounds__(256)
void cvt_f32_to_f16(const float* __restrict__ in, _Float16* __restrict__ out, int n4) {
    int i = blockIdx.x * blockDim.x + threadIdx.x;
    if (i < n4) {
        float4 v = ((const float4*)in)[i];
        v4h h;
        h[0] = (_Float16)v.x;
        h[1] = (_Float16)v.y;
        h[2] = (_Float16)v.z;
        h[3] = (_Float16)v.w;
        ((v4h*)out)[i] = h;
    }
}

// ---------------------------------------------------------------------------
// Kernel 2: fused convert + transpose for W:  WT[n][k] = (f16)W[k][n]
// 64x64 tiles staged through LDS so both global phases are coalesced.
// ---------------------------------------------------------------------------
__global__ __launch_bounds__(256)
void cvt_transpose_w(const float* __restrict__ W, _Float16* __restrict__ WT) {
    __shared__ _Float16 tile[64][72];            // [k][n], padded stride
    const int k0 = blockIdx.y * 64;
    const int n0 = blockIdx.x * 64;
    const int tid = threadIdx.x;
    const int seg = tid & 15;                    // 4-element column segment
    const int row = tid >> 4;                    // 0..15

#pragma unroll
    for (int it = 0; it < 4; ++it) {
        int k = row + it * 16;
        float4 v = *(const float4*)(W + (size_t)(k0 + k) * OUT_SZ + n0 + seg * 4);
        v4h h;
        h[0] = (_Float16)v.x; h[1] = (_Float16)v.y;
        h[2] = (_Float16)v.z; h[3] = (_Float16)v.w;
        *(v4h*)&tile[k][seg * 4] = h;
    }
    __syncthreads();
#pragma unroll
    for (int it = 0; it < 4; ++it) {
        int n = row + it * 16;
        int kk = seg * 4;
        v4h h;
#pragma unroll
        for (int i = 0; i < 4; ++i) h[i] = tile[kk + i][n];
        *(v4h*)(WT + (size_t)(n0 + n) * IN_SZ + k0 + kk) = h;
    }
}

// ---------------------------------------------------------------------------
// Kernel 3: GEMM  Y[4096,2048] = Xh[4096,2048] @ WT^T   (f16 -> f32)
// 256 thr = 8 waves (2 M x 4 N); WG tile M=64,N=256; wave tile 32x64.
// Double-buffered LDS, one barrier per K-chunk of 32.
// ---------------------------------------------------------------------------
__device__ __forceinline__ v16h cat8(v8h lo, v8h hi) {
    v16h r;
#pragma unroll
    for (int i = 0; i < 8; ++i) { r[i] = lo[i]; r[i + 8] = hi[i]; }
    return r;
}

__global__ __launch_bounds__(256)
void snn_gemm(const _Float16* __restrict__ A,    // [4096, 2048] row-major f16
              const _Float16* __restrict__ BT,   // [2048(N), 2048(K)] f16
              float* __restrict__ Y) {           // [4096, 2048] f32
    __shared__ __align__(16) _Float16 As[2][64][32];     //  8 KB
    __shared__ __align__(16) _Float16 Bs[2][256][32];    // 32 KB

    const int m0 = blockIdx.y * 64;
    const int n0 = blockIdx.x * 256;
    const int tid  = threadIdx.x;
    const int lane = tid & 31;
    const int wave = tid >> 5;            // 0..7
    const int wm   = wave >> 2;           // 0..1  -> M offset wm*32
    const int wn   = wave & 3;            // 0..3  -> N offset wn*64
    const int g    = lane >> 4;           // half-wave id
    const int l16  = lane & 15;

    // staging roles (contiguous b128 on both sides now)
    const int ar = tid >> 2, ac = tid & 3;                      // A: 64 rows x 4 segs
    const _Float16* aSrc = A  + (size_t)(m0 + ar) * IN_SZ + ac * 8;
    const _Float16* bSrc = BT + (size_t)(n0 + tid) * IN_SZ;     // one n-row / thread

    v8f acc[2][4];
#pragma unroll
    for (int i = 0; i < 2; ++i)
#pragma unroll
        for (int j = 0; j < 4; ++j) acc[i][j] = (v8f)(0.0f);

    // prologue: stage K-chunk 0 into buffer 0
    {
        v8h a0 = *(const v8h*)(aSrc);
        *(v8h*)&As[0][ar][ac * 8] = a0;
#pragma unroll
        for (int s = 0; s < 4; ++s) {
            v8h b0 = *(const v8h*)(bSrc + s * 8);
            *(v8h*)&Bs[0][tid][s * 8] = b0;
        }
    }
    __syncthreads();

    const int NK = IN_SZ / 32;            // 64 K-chunks
    for (int ki = 0; ki < NK; ++ki) {
        const int cur = ki & 1;
        const int nxt = cur ^ 1;
        const bool hasNext = (ki + 1) < NK;

        // prefetch next K-chunk into registers (latency hidden by WMMAs)
        v8h aReg;
        v8h bReg[4];
        if (hasNext) {
            const int kc = (ki + 1) * 32;
            aReg = *(const v8h*)(aSrc + kc);
#pragma unroll
            for (int s = 0; s < 4; ++s) bReg[s] = *(const v8h*)(bSrc + kc + s * 8);
        }

        // fragments from buffer `cur` per documented 16-bit WMMA layouts
        v16h afrag[2];
#pragma unroll
        for (int i = 0; i < 2; ++i) {
            const int row = wm * 32 + i * 16 + l16;
            v8h lo = *(const v8h*)&As[cur][row][g * 8];        // K = g*8 .. +7
            v8h hi = *(const v8h*)&As[cur][row][16 + g * 8];   // K = 16+g*8 .. +7
            afrag[i] = cat8(lo, hi);
        }
#pragma unroll
        for (int j = 0; j < 4; ++j) {
            const int nn = wn * 64 + j * 16 + l16;
            v8h lo = *(const v8h*)&Bs[cur][nn][g * 8];         // K = g*8 .. +7
            v8h hi = *(const v8h*)&Bs[cur][nn][16 + g * 8];    // K = 16+g*8 .. +7
            v16h bfrag = cat8(lo, hi);
#pragma unroll
            for (int i = 0; i < 2; ++i)
                acc[i][j] = __builtin_amdgcn_wmma_f32_16x16x32_f16(
                    /*neg_a=*/false, afrag[i],
                    /*neg_b=*/false, bfrag,
                    /*c_mod=*/(short)0, acc[i][j],
                    /*reuse_a=*/false, /*reuse_b=*/false);
        }

        // stage next chunk into the other buffer, then single barrier
        if (hasNext) {
            *(v8h*)&As[nxt][ar][ac * 8] = aReg;
#pragma unroll
            for (int s = 0; s < 4; ++s) *(v8h*)&Bs[nxt][tid][s * 8] = bReg[s];
        }
        __syncthreads();
    }

    // ---- store C:  VGPR r -> row (g*8 + r), col = l16 ----
#pragma unroll
    for (int i = 0; i < 2; ++i) {
#pragma unroll
        for (int j = 0; j < 4; ++j) {
            const int col = n0 + wn * 64 + j * 16 + l16;
#pragma unroll
            for (int r = 0; r < 8; ++r) {
                const int row = m0 + wm * 32 + i * 16 + g * 8 + r;
                Y[(size_t)row * OUT_SZ + col] = acc[i][j][r];
            }
        }
    }
}

// ---------------------------------------------------------------------------
// Kernel 4: leaky-integrate / spike / soft-reset scan over T (state in regs)
// One thread per (b, o) pair; coalesced over o; NT stores for spikes.
// ---------------------------------------------------------------------------
__global__ __launch_bounds__(256)
void snn_scan(const float* __restrict__ Y, float* __restrict__ out) {
    int tid = blockIdx.x * blockDim.x + threadIdx.x;   // 0 .. B*OUT-1
    size_t base   = (size_t)tid;                       // = b*OUT + o
    size_t stride = (size_t)BATCH * OUT_SZ;            // per-t stride
    float u = 0.0f;
#pragma unroll
    for (int t = 0; t < T_STEPS; ++t) {
        float y = Y[base + (size_t)t * stride];
        u = SIG_TAU * u + y;
        float s = (u >= 1.0f) ? 1.0f : 0.0f;
        __builtin_nontemporal_store(s, &out[base + (size_t)t * stride]);
        u -= s;
    }
}

// ---------------------------------------------------------------------------
extern "C" void kernel_launch(void* const* d_in, const int* in_sizes, int n_in,
                              void* d_out, int out_size, void* d_ws, size_t ws_size,
                              hipStream_t stream) {
    const float* x = (const float*)d_in[0];   // [32,128,2048] f32
    const float* w = (const float*)d_in[1];   // [2048,2048]   f32
    float* out = (float*)d_out;               // [32,128,2048] f32

    char* ws = (char*)d_ws;
    _Float16* Xh = (_Float16*)(ws);                       // 16 MiB
    _Float16* WT = (_Float16*)(ws + (16u << 20));         //  8 MiB  (transposed W)
    float*    Y  = (float*)   (ws + (24u << 20));         // 32 MiB

    const int nX = T_STEPS * BATCH * IN_SZ;   // 8388608

    cvt_f32_to_f16<<<(nX / 4 + 255) / 256, 256, 0, stream>>>(x, Xh, nX / 4);

    dim3 tgrid(OUT_SZ / 64, IN_SZ / 64);      // (32, 32)
    cvt_transpose_w<<<tgrid, 256, 0, stream>>>(w, WT);

    dim3 grid(OUT_SZ / 256, M_TOT / 64);      // (8, 64)
    snn_gemm<<<grid, 256, 0, stream>>>(Xh, WT, Y);

    snn_scan<<<(BATCH * OUT_SZ) / 256, 256, 0, stream>>>(Y, out);
}
